// CrossBatchAttention_89309549953246
// MI455X (gfx1250) — matrix operand order repaired and checked by reading, not compiled, not run on hardware
//
#include <hip/hip_runtime.h>
#include <hip/hip_bf16.h>
#include <math.h>

typedef __bf16 bf16;
typedef __attribute__((ext_vector_type(8)))  bf16  bf16x8;
typedef __attribute__((ext_vector_type(16))) bf16  bf16x16;
typedef __attribute__((ext_vector_type(8)))  float floatx8;
typedef __attribute__((ext_vector_type(4)))  float floatx4;

#define HIDDEN      4096
#define NUM_HEADS   32
#define HEAD_DIM    128
#define GATE_HIDDEN 1024
#define BATCH       2048

#if __has_builtin(__builtin_amdgcn_global_load_async_to_lds_b128) && \
    __has_builtin(__builtin_amdgcn_s_wait_asynccnt)
#define USE_ASYNC 1
#else
#define USE_ASYNC 0
#endif

// GCC-style vector to match the builtin's parameter type exactly
typedef int v4i __attribute__((vector_size(16)));
typedef __attribute__((address_space(1))) v4i g_v4i;   // global (AS1)
typedef __attribute__((address_space(3))) v4i l_v4i;   // LDS (AS3)

static __device__ __forceinline__ bf16x16 cat8(bf16x8 lo, bf16x8 hi) {
    return __builtin_shufflevector(lo, hi, 0,1,2,3,4,5,6,7,8,9,10,11,12,13,14,15);
}

static __device__ __forceinline__ void async_copy16(const void* g, void* l) {
#if USE_ASYNC
    __builtin_amdgcn_global_load_async_to_lds_b128(
        (g_v4i*)(uintptr_t)g,
        (l_v4i*)(unsigned)(uintptr_t)l,   // low 32 bits of flat LDS addr = LDS offset
        0, 0);
#else
    *(bf16x8*)l = *(const bf16x8*)g;
#endif
}

static __device__ __forceinline__ void async_wait0() {
#if USE_ASYNC
    __builtin_amdgcn_s_wait_asynccnt(0);
#endif
}

// ---------------------------------------------------------------------------
// bf16 WMMA GEMM:  C[M,N] = act(A[M,K] * Bt[N,K]^T + bias)
// A bf16 [M][K] row-major; Bt bf16 [N][K] (pre-transposed weight) -> both tile
// loaders are pure 16B copies (async global->LDS when available).
// Block 128x128, BK=64, double-buffered LDS. 8 waves (2x4), each 64x32.
// act: 0=none, 1=gelu(exact), 2=sigmoid. Output Cb (bf16) if non-null else Cf.
// ---------------------------------------------------------------------------
#define BM  128
#define BN  128
#define BK  64
#define LDT 72   // 64 + 8 pad bf16 elems -> 144B row, 16B aligned, bank-spread

static __device__ __forceinline__ void copy_tile(bf16* dst, const bf16* src,
                                                 unsigned row0, unsigned k0,
                                                 unsigned K, unsigned tid)
{
#if USE_ASYNC
#pragma unroll
    for (int i = 0; i < 4; ++i) {
        unsigned c = tid + i * 256, r = c >> 3, ch = (c & 7) * 8;
        async_copy16(src + (row0 + r) * K + k0 + ch, dst + r * LDT + ch);
    }
#else
    bf16x8 tmp[4]; unsigned rr[4], cc[4];
#pragma unroll
    for (int i = 0; i < 4; ++i) {
        unsigned c = tid + i * 256; rr[i] = c >> 3; cc[i] = (c & 7) * 8;
        tmp[i] = *(const bf16x8*)(src + (row0 + rr[i]) * K + k0 + cc[i]);
    }
#pragma unroll
    for (int i = 0; i < 4; ++i)
        *(bf16x8*)(dst + rr[i] * LDT + cc[i]) = tmp[i];
#endif
}

__global__ __launch_bounds__(256)
void gemm_bf16_wmma(const bf16* __restrict__ A, const bf16* __restrict__ Bt,
                    const float* __restrict__ bias,
                    float* __restrict__ Cf, bf16* __restrict__ Cb,
                    int M, int N, int K, int act)
{
    __shared__ bf16 As[2][BM * LDT];
    __shared__ bf16 Bs[2][BM * LDT];

    const unsigned tid   = threadIdx.x;
    const unsigned lane  = tid & 31;
    const unsigned wave  = tid >> 5;
    const unsigned l16   = lane & 15;
    const unsigned lhalf = lane >> 4;
    const unsigned waveM = wave >> 2;    // 0..1
    const unsigned waveN = wave & 3;     // 0..3
    const unsigned bRow  = blockIdx.y * BM;
    const unsigned bCol  = blockIdx.x * BN;

    floatx8 acc[4][2];
#pragma unroll
    for (int i = 0; i < 4; ++i)
#pragma unroll
        for (int j = 0; j < 2; ++j) acc[i][j] = (floatx8){};

    copy_tile(As[0], A,  bRow, 0, K, tid);
    copy_tile(Bs[0], Bt, bCol, 0, K, tid);
    async_wait0();
    __syncthreads();

    const int nt = K / BK;
    for (int t = 0; t < nt; ++t) {
        const int b = t & 1;
        if (t + 1 < nt) {                       // prefetch next tile pair
            copy_tile(As[b ^ 1], A,  bRow, (unsigned)(t + 1) * BK, K, tid);
            copy_tile(Bs[b ^ 1], Bt, bCol, (unsigned)(t + 1) * BK, K, tid);
        }
#pragma unroll
        for (int ks = 0; ks < 2; ++ks) {        // two K=32 steps per tile
            bf16x16 af[4];
#pragma unroll
            for (int tm = 0; tm < 4; ++tm) {
                unsigned row = waveM * 64 + tm * 16 + l16;
                const bf16* p = &As[b][row * LDT + ks * 32 + lhalf * 8];
                af[tm] = cat8(*(const bf16x8*)p, *(const bf16x8*)(p + 16));
            }
            bf16x16 bfg[2];
#pragma unroll
            for (int tn = 0; tn < 2; ++tn) {
                unsigned n = waveN * 32 + tn * 16 + l16;
                const bf16* p = &Bs[b][n * LDT + ks * 32 + lhalf * 8];
                bfg[tn] = cat8(*(const bf16x8*)p, *(const bf16x8*)(p + 16));
            }
#pragma unroll
            for (int tm = 0; tm < 4; ++tm)
#pragma unroll
                for (int tn = 0; tn < 2; ++tn)
                    acc[tm][tn] = __builtin_amdgcn_wmma_f32_16x16x32_bf16(
                        false, af[tm], false, bfg[tn], (short)0, acc[tm][tn],
                        false, false);
        }
        async_wait0();
        __syncthreads();
    }

    // epilogue: C layout M = v + 8*lhalf, N = l16
#pragma unroll
    for (int tm = 0; tm < 4; ++tm) {
#pragma unroll
        for (int tn = 0; tn < 2; ++tn) {
            unsigned col = bCol + waveN * 32 + tn * 16 + l16;
            float bv = bias ? bias[col] : 0.0f;
#pragma unroll
            for (int v = 0; v < 8; ++v) {
                unsigned row = bRow + waveM * 64 + tm * 16 + lhalf * 8 + v;
                float x = acc[tm][tn][v] + bv;
                if (act == 1)      x = 0.5f * x * (1.0f + erff(x * 0.70710678118654752f));
                else if (act == 2) x = 1.0f / (1.0f + __expf(-x));
                unsigned idx = row * (unsigned)N + col;
                if (Cb) Cb[idx] = (bf16)x;
                else    Cf[idx] = x;
            }
        }
    }
}

// ---------------------------------------------------------------------------
// Flash-style cross-batch attention. Grid: (BATCH/128, NUM_HEADS), 256 thr.
// ---------------------------------------------------------------------------
__global__ __launch_bounds__(256)
void flash_attn_wmma(const bf16* __restrict__ Qb, const bf16* __restrict__ Kb,
                     const bf16* __restrict__ Vb,
                     const unsigned char* __restrict__ mask,
                     bf16* __restrict__ Out)
{
    __shared__ bf16 Vt[128 * 40];        // [d][key] transposed V block
    __shared__ bf16 Ps[8 * 16 * 40];     // per-wave P staging

    const unsigned tid   = threadIdx.x;
    const unsigned lane  = tid & 31;
    const unsigned wave  = tid >> 5;
    const unsigned l16   = lane & 15;
    const unsigned lhalf = lane >> 4;
    const unsigned head  = blockIdx.y;
    const unsigned hoff  = head * HEAD_DIM;
    const unsigned r0    = blockIdx.x * 128 + wave * 16;

    bf16x16 qf[4];
    {
        const bf16* qrow = Qb + (r0 + l16) * HIDDEN + hoff;
#pragma unroll
        for (int kc = 0; kc < 4; ++kc) {
            const bf16* p = qrow + kc * 32 + lhalf * 8;
            qf[kc] = cat8(*(const bf16x8*)p, *(const bf16x8*)(p + 16));
        }
    }

    float mi[8], li[8];
    floatx8 o[8];
#pragma unroll
    for (int v = 0; v < 8; ++v) { mi[v] = -1e30f; li[v] = 0.0f; }
#pragma unroll
    for (int dt = 0; dt < 8; ++dt) o[dt] = (floatx8){};

    const float scale = 0.08838834764831845f;   // 1/sqrt(128)

    for (unsigned key0 = 0; key0 < BATCH; key0 += 32) {
        __syncthreads();
        // stage V block [32 keys][128 d] transposed (batch loads, then scatter)
        {
            unsigned c0 = tid, c1 = tid + 256;
            unsigned kk0 = c0 >> 4, d80 = (c0 & 15) * 8;
            unsigned kk1 = c1 >> 4, d81 = (c1 & 15) * 8;
            bf16x8 v0 = *(const bf16x8*)(Vb + (key0 + kk0) * HIDDEN + hoff + d80);
            bf16x8 v1 = *(const bf16x8*)(Vb + (key0 + kk1) * HIDDEN + hoff + d81);
#pragma unroll
            for (int j = 0; j < 8; ++j) Vt[(d80 + j) * 40 + kk0] = v0[j];
#pragma unroll
            for (int j = 0; j < 8; ++j) Vt[(d81 + j) * 40 + kk1] = v1[j];
        }
        __syncthreads();

        // S = Q K^T (K fragments straight from global: two 8-d runs per lane, 16 apart)
        floatx8 sacc[2] = {(floatx8){}, (floatx8){}};
#pragma unroll
        for (int tn = 0; tn < 2; ++tn) {
            const bf16* krow = Kb + (key0 + tn * 16 + l16) * HIDDEN + hoff;
#pragma unroll
            for (int kc = 0; kc < 4; ++kc) {
                const bf16* p = krow + kc * 32 + lhalf * 8;
                bf16x16 kf = cat8(*(const bf16x8*)p, *(const bf16x8*)(p + 16));
                sacc[tn] = __builtin_amdgcn_wmma_f32_16x16x32_bf16(
                    false, qf[kc], false, kf, (short)0, sacc[tn], false, false);
            }
        }

        // scale, diag+mask exclusion, online softmax (half-wave reductions)
        float pv0[8], pv1[8];
#pragma unroll
        for (int v = 0; v < 8; ++v) {
            int row_g = (int)(r0 + lhalf * 8 + v);
            int c0 = (int)(key0 + l16), c1 = (int)(key0 + 16 + l16);
            float s0 = sacc[0][v] * scale;
            float s1 = sacc[1][v] * scale;
            if (row_g == c0 || !mask[c0]) s0 = -1e30f;
            if (row_g == c1 || !mask[c1]) s1 = -1e30f;
            float sm = fmaxf(s0, s1);
            sm = fmaxf(sm, __shfl_xor(sm, 8, 32));
            sm = fmaxf(sm, __shfl_xor(sm, 4, 32));
            sm = fmaxf(sm, __shfl_xor(sm, 2, 32));
            sm = fmaxf(sm, __shfl_xor(sm, 1, 32));
            float mnew  = fmaxf(mi[v], sm);
            float alpha = __expf(mi[v] - mnew);
            float p0 = __expf(s0 - mnew);
            float p1 = __expf(s1 - mnew);
            float ps = p0 + p1;
            ps += __shfl_xor(ps, 8, 32);
            ps += __shfl_xor(ps, 4, 32);
            ps += __shfl_xor(ps, 2, 32);
            ps += __shfl_xor(ps, 1, 32);
            li[v] = li[v] * alpha + ps;
            mi[v] = mnew;
#pragma unroll
            for (int dt = 0; dt < 8; ++dt) o[dt][v] *= alpha;
            pv0[v] = p0; pv1[v] = p1;
        }

        // C-layout -> A-layout for P via per-wave LDS region
        bf16* pw = &Ps[wave * 16 * 40];
#pragma unroll
        for (int v = 0; v < 8; ++v) {
            pw[(lhalf * 8 + v) * 40 + l16]      = (bf16)pv0[v];
            pw[(lhalf * 8 + v) * 40 + 16 + l16] = (bf16)pv1[v];
        }
        asm volatile("s_wait_dscnt 0" ::: "memory");   // intra-wave DS RAW fence

        bf16x16 pf;
        {
            const bf16* p = pw + l16 * 40 + lhalf * 8;
            pf = cat8(*(const bf16x8*)p, *(const bf16x8*)(p + 16));
        }

        // O += P V
#pragma unroll
        for (int dt = 0; dt < 8; ++dt) {
            const bf16* p = &Vt[(dt * 16 + l16) * 40 + lhalf * 8];
            bf16x16 vf = cat8(*(const bf16x8*)p, *(const bf16x8*)(p + 16));
            o[dt] = __builtin_amdgcn_wmma_f32_16x16x32_bf16(
                false, pf, false, vf, (short)0, o[dt], false, false);
        }
    }

#pragma unroll
    for (int v = 0; v < 8; ++v) {
        float inv = li[v] > 0.0f ? 1.0f / li[v] : 0.0f;   // nan_to_num
        unsigned row = r0 + lhalf * 8 + v;
#pragma unroll
        for (int dt = 0; dt < 8; ++dt)
            Out[row * HIDDEN + hoff + dt * 16 + l16] = (bf16)(o[dt][v] * inv);
    }
}

// ---------------------------------------------------------------------------
// W fp32 [K][N]  ->  Wt bf16 [N][K]   (64x64 tiles via LDS)
// ---------------------------------------------------------------------------
__global__ __launch_bounds__(256)
void transpose_cvt(const float* __restrict__ W, bf16* __restrict__ Wt,
                   int K, int N)
{
    __shared__ bf16 t[64 * 72];
    const unsigned tid = threadIdx.x;
    const unsigned n0 = blockIdx.x * 64, k0 = blockIdx.y * 64;
#pragma unroll
    for (int i = 0; i < 4; ++i) {                 // 1024 float4 chunks
        unsigned c = tid + i * 256, r = c >> 4, col = (c & 15) * 4;
        floatx4 v = *(const floatx4*)(W + (k0 + r) * (unsigned)N + n0 + col);
        bf16* d = &t[r * 72 + col];
        d[0] = (bf16)v[0]; d[1] = (bf16)v[1]; d[2] = (bf16)v[2]; d[3] = (bf16)v[3];
    }
    __syncthreads();
#pragma unroll
    for (int i = 0; i < 2; ++i) {                 // 512 bf16x8 out chunks
        unsigned c = tid + i * 256, nr = c >> 3, kc = (c & 7) * 8;
        bf16x8 o;
#pragma unroll
        for (int j = 0; j < 8; ++j) o[j] = t[(kc + j) * 72 + nr];
        *(bf16x8*)(Wt + (n0 + nr) * (unsigned)K + k0 + kc) = o;
    }
}

// ---------------------------------------------------------------------------
__global__ __launch_bounds__(256)
void cvt_bf16_kernel(const float* __restrict__ in, bf16* __restrict__ out)
{
    unsigned i = (blockIdx.x * 256 + threadIdx.x) * 4;
    floatx4 v = *(const floatx4*)(in + i);
    bf16* o = out + i;
    o[0] = (bf16)v[0]; o[1] = (bf16)v[1]; o[2] = (bf16)v[2]; o[3] = (bf16)v[3];
}

// gate_in = concat([X, cross]) as bf16  (2048 x 8192)
__global__ __launch_bounds__(256)
void concat_bf16_kernel(const float* __restrict__ X, const float* __restrict__ C,
                        bf16* __restrict__ G)
{
    unsigned i = (blockIdx.x * 256 + threadIdx.x) * 4;
    unsigned row = i >> 13, col = i & 8191;
    const float* src = (col < HIDDEN) ? X + row * HIDDEN + col
                                      : C + row * HIDDEN + (col - HIDDEN);
    floatx4 v = *(const floatx4*)src;
    bf16* o = G + i;
    o[0] = (bf16)v[0]; o[1] = (bf16)v[1]; o[2] = (bf16)v[2]; o[3] = (bf16)v[3];
}

// out = X + gate * cross
__global__ __launch_bounds__(256)
void combine_kernel(const float* __restrict__ X, const float* __restrict__ gate,
                    const float* __restrict__ cross, float* __restrict__ out)
{
    unsigned i = (blockIdx.x * 256 + threadIdx.x) * 4;
    floatx4 x = *(const floatx4*)(X + i);
    floatx4 g = *(const floatx4*)(gate + i);
    floatx4 c = *(const floatx4*)(cross + i);
    floatx4 r;
#pragma unroll
    for (int j = 0; j < 4; ++j) r[j] = x[j] + g[j] * c[j];
    *(floatx4*)(out + i) = r;
}

// ---------------------------------------------------------------------------
extern "C" void kernel_launch(void* const* d_in, const int* in_sizes, int n_in,
                              void* d_out, int out_size, void* d_ws, size_t ws_size,
                              hipStream_t stream)
{
    (void)in_sizes; (void)n_in; (void)out_size; (void)ws_size;
    const float* X    = (const float*)d_in[0];
    const unsigned char* mask = (const unsigned char*)d_in[1];
    const float* Wq   = (const float*)d_in[2];
    const float* Wk   = (const float*)d_in[3];
    const float* Wv   = (const float*)d_in[4];
    const float* Wo   = (const float*)d_in[5];
    const float* gW1  = (const float*)d_in[6];
    const float* gb1  = (const float*)d_in[7];
    const float* gW2  = (const float*)d_in[8];
    const float* gb2  = (const float*)d_in[9];
    float* out = (float*)d_out;

    const size_t MB = (size_t)1 << 20;
    char* ws = (char*)d_ws;
    bf16*  Wqt    = (bf16*) (ws +   0 * MB);   // [4096][4096] 32 MB
    bf16*  Wkt    = (bf16*) (ws +  32 * MB);   // 32 MB
    bf16*  Wvt    = (bf16*) (ws +  64 * MB);   // 32 MB
    bf16*  Wot    = (bf16*) (ws +  96 * MB);   // 32 MB
    bf16*  gW1t   = (bf16*) (ws + 128 * MB);   // [1024][8192] 16 MB
    bf16*  gW2t   = (bf16*) (ws + 144 * MB);   // [4096][1024]  8 MB
    bf16*  Xb     = (bf16*) (ws + 152 * MB);   // 16 MB
    bf16*  Qb     = (bf16*) (ws + 168 * MB);   // 16 MB
    bf16*  Kb     = (bf16*) (ws + 184 * MB);   // 16 MB
    bf16*  Vb     = (bf16*) (ws + 200 * MB);   // 16 MB
    bf16*  attnb  = (bf16*) (ws + 216 * MB);   // 16 MB
    float* cross  = (float*)(ws + 232 * MB);   // 32 MB
    bf16*  gateinb= (bf16*) (ws + 264 * MB);   // 32 MB
    bf16*  g1b    = (bf16*) (ws + 296 * MB);   //  4 MB
    float* gate   = (float*)(ws + 300 * MB);   // 32 MB

    dim3 blk(256);

    // prologue: weights -> bf16 [N][K]; X -> bf16
    transpose_cvt<<<dim3(HIDDEN / 64, HIDDEN / 64), blk, 0, stream>>>(Wq, Wqt, HIDDEN, HIDDEN);
    transpose_cvt<<<dim3(HIDDEN / 64, HIDDEN / 64), blk, 0, stream>>>(Wk, Wkt, HIDDEN, HIDDEN);
    transpose_cvt<<<dim3(HIDDEN / 64, HIDDEN / 64), blk, 0, stream>>>(Wv, Wvt, HIDDEN, HIDDEN);
    transpose_cvt<<<dim3(HIDDEN / 64, HIDDEN / 64), blk, 0, stream>>>(Wo, Wot, HIDDEN, HIDDEN);
    transpose_cvt<<<dim3(GATE_HIDDEN / 64, (2 * HIDDEN) / 64), blk, 0, stream>>>(gW1, gW1t, 2 * HIDDEN, GATE_HIDDEN);
    transpose_cvt<<<dim3(HIDDEN / 64, GATE_HIDDEN / 64), blk, 0, stream>>>(gW2, gW2t, GATE_HIDDEN, HIDDEN);
    cvt_bf16_kernel<<<(BATCH * HIDDEN) / 1024, blk, 0, stream>>>(X, Xb);

    dim3 gHH(HIDDEN / BN, BATCH / BM);          // (32, 16)
    dim3 gG1(GATE_HIDDEN / BN, BATCH / BM);     // (8, 16)

    gemm_bf16_wmma<<<gHH, blk, 0, stream>>>(Xb, Wqt, nullptr, nullptr, Qb,
                                            BATCH, HIDDEN, HIDDEN, 0);
    gemm_bf16_wmma<<<gHH, blk, 0, stream>>>(Xb, Wkt, nullptr, nullptr, Kb,
                                            BATCH, HIDDEN, HIDDEN, 0);
    gemm_bf16_wmma<<<gHH, blk, 0, stream>>>(Xb, Wvt, nullptr, nullptr, Vb,
                                            BATCH, HIDDEN, HIDDEN, 0);

    flash_attn_wmma<<<dim3(BATCH / 128, NUM_HEADS), blk, 0, stream>>>(
        Qb, Kb, Vb, mask, attnb);

    gemm_bf16_wmma<<<gHH, blk, 0, stream>>>(attnb, Wot, nullptr, cross, nullptr,
                                            BATCH, HIDDEN, HIDDEN, 0);

    concat_bf16_kernel<<<(BATCH * 2 * HIDDEN) / 1024, blk, 0, stream>>>(X, cross, gateinb);
    gemm_bf16_wmma<<<gG1, blk, 0, stream>>>(gateinb, gW1t, gb1, nullptr, g1b,
                                            BATCH, GATE_HIDDEN, 2 * HIDDEN, 1);
    gemm_bf16_wmma<<<gHH, blk, 0, stream>>>(g1b, gW2t, gb2, gate, nullptr,
                                            BATCH, HIDDEN, GATE_HIDDEN, 2);

    combine_kernel<<<(BATCH * HIDDEN) / 1024, blk, 0, stream>>>(X, gate, cross, out);
}
